// TokenEmbedding_84524956385735
// MI455X (gfx1250) — compile-verified, matrix operand order checked by reading, not run on hardware
//
#include <hip/hip_runtime.h>

// ---------------- problem constants (derived from the reference) -------------
#define B_     16
#define S_     2048
#define C_     32
#define PAD_   21
#define SIGLEN 2080           // per-channel padded signal length in LDS (>=2072)
#define CPB    4              // channels staged per block
#define NCG    (C_ / CPB)     // channel groups per batch
#define HMAX   680
#define OUTPB  (S_ * 74)      // 151552 output elements per batch (the kept slice)

typedef __attribute__((ext_vector_type(2)))  float    v2f;
typedef __attribute__((ext_vector_type(8)))  float    v8f;
typedef __attribute__((ext_vector_type(16))) _Float16 v16h;

// ---- feature detection (device pass only; host pass takes scalar tier) ------
#if defined(__HIP_DEVICE_COMPILE__) && __has_builtin(__builtin_amdgcn_wmma_f32_16x16x4_f32)
#define WMMA_PATH 2   // exact fp32 WMMA, K chunks of 4
#elif defined(__HIP_DEVICE_COMPILE__) && __has_builtin(__builtin_amdgcn_wmma_f32_16x16x32_f16)
#define WMMA_PATH 1   // f16 inputs, fp32 accumulate, single K=32 WMMA
#else
#define WMMA_PATH 0   // scalar (host-compile tier)
#endif

#if defined(__HIP_DEVICE_COMPILE__) && __has_builtin(__builtin_amdgcn_global_load_async_to_lds_b32)
#define ASYNC_LDS 1
#else
#define ASYNC_LDS 0
#endif

__global__ __launch_bounds__(256) void
conv24_wmma_kernel(const float* __restrict__ x,
                   const float* __restrict__ kern,
                   float* __restrict__ out)
{
    extern __shared__ float sig[];            // [CPB][SIGLEN]
    const int b     = blockIdx.x >> 3;        // batch
    const int cbase = (blockIdx.x & (NCG - 1)) * CPB;
    const int tid   = threadIdx.x;

    // ---- zero the pad regions (front 21, tail 2069..2079) -------------------
    if (tid < CPB * 32) {
        int cl = tid >> 5, j = tid & 31;
        if (j < PAD_) sig[cl * SIGLEN + j] = 0.0f;
        int tj = PAD_ + S_ + j;
        if (tj < SIGLEN) sig[cl * SIGLEN + tj] = 0.0f;
    }

    // ---- stage x[b, :, cbase..cbase+3] transposed into LDS ------------------
    {
        const float* gsrc = x + ((size_t)b * S_) * C_ + cbase;
        for (int e = tid; e < S_ * CPB; e += 256) {
            int s  = e >> 2;
            int cl = e & (CPB - 1);
            const float* gp = gsrc + (size_t)s * C_ + cl;
            float*       lp = &sig[cl * SIGLEN + PAD_ + s];
#if ASYNC_LDS
            __builtin_amdgcn_global_load_async_to_lds_b32(
                (__attribute__((address_space(1))) int*)gp,
                (__attribute__((address_space(3))) int*)lp, 0, 0);
#else
            *lp = *gp;
#endif
        }
    }
#if ASYNC_LDS
#if __has_builtin(__builtin_amdgcn_s_wait_asynccnt)
    __builtin_amdgcn_s_wait_asynccnt(0);
#else
    asm volatile("s_wait_asynccnt 0" ::: "memory");
#endif
#endif
    __syncthreads();

    // ---- per-lane constants -------------------------------------------------
    const int lane = tid & 31;
    const int col  = lane & 15;     // D/B column, also A row index M
    const int hi2  = lane >> 4;     // half-wave selector
    const int sft  = col >> 1;      // shift s in 0..7
    const int nk   = col & 1;       // filter index 0/1

#if WMMA_PATH == 2
    // B[d, col] = K24[nk, d - s] (zero-padded), packed for 16x16x4 fp32 WMMA:
    // chunk k, lane half hi2 -> taps d = 4k + 2*hi2 + {0,1}
    v2f bfrag[8];
#pragma unroll
    for (int k = 0; k < 8; ++k) {
        int d0 = 4 * k + 2 * hi2;
        int e0 = d0 - sft, e1 = d0 + 1 - sft;
        bfrag[k].x = (e0 >= 0 && e0 < 24) ? kern[nk * 24 + e0] : 0.0f;
        bfrag[k].y = (e1 >= 0 && e1 < 24) ? kern[nk * 24 + e1] : 0.0f;
    }
#elif WMMA_PATH == 1
    v16h bh;
#pragma unroll
    for (int v = 0; v < 8; ++v)
#pragma unroll
        for (int p = 0; p < 2; ++p) {
            int K = (v >> 2) * 16 + hi2 * 8 + 2 * (v & 3) + p;
            int e = K - sft;
            bh[2 * v + p] = (_Float16)((e >= 0 && e < 24) ? kern[nk * 24 + e] : 0.0f);
        }
#endif

    // ---- wave-uniform tile loop: CPB*16 tiles of 128 t-values each ----------
    const int wave = tid >> 5;
    const long obase = (long)b * OUTPB;
    for (int tile = wave; tile < CPB * 16; tile += 8) {
        const int   cl = tile >> 4;
        const int   T0 = (tile & 15) << 7;       // t base, 0..1920
        const float* sg = &sig[cl * SIGLEN];
        const int   arow = T0 + 8 * col;         // A row M = col
        v8f acc = {};

#if WMMA_PATH == 2
#pragma unroll
        for (int k = 0; k < 8; ++k) {
            int base = arow + 4 * k + 2 * hi2;
            v2f a;
            a.x = sg[base];
            a.y = sg[base + 1];
            acc = __builtin_amdgcn_wmma_f32_16x16x4_f32(
                      false, a, false, bfrag[k], (short)0, acc, false, false);
        }
#elif WMMA_PATH == 1
        v16h ah;
#pragma unroll
        for (int v = 0; v < 8; ++v)
#pragma unroll
            for (int p = 0; p < 2; ++p) {
                int K = (v >> 2) * 16 + hi2 * 8 + 2 * (v & 3) + p;
                ah[2 * v + p] = (_Float16)sg[arow + K];
            }
        acc = __builtin_amdgcn_wmma_f32_16x16x32_f16(
                  false, ah, false, bh, (short)0, acc, false, false);
#else
        // scalar tier (host-compile / no-WMMA fallback)
#pragma unroll
        for (int i = 0; i < 8; ++i) {
            int r = i + 8 * hi2;
            int t = T0 + 8 * r + sft;
            float y = 0.0f;
            for (int e = 0; e < 24; ++e) y += sg[t + e] * kern[nk * 24 + e];
            acc[i] = y;
        }
#endif

        // ---- scatter D: lane holds y(nk, t) for rows r = i + 8*hi2 ---------
        const int cj = nk * 32 + (cbase + cl);
#pragma unroll
        for (int i = 0; i < 8; ++i) {
            int   r = i + 8 * hi2;
            int   t = T0 + 8 * r + sft;
            float y = acc[i];
            int h1 = t / 3, w1 = t - 3 * h1;
            if (h1 <= HMAX) {
                int pos = (cj * 681 + h1) * 6 + w1;
                if (pos < OUTPB) out[obase + pos] = y;
            }
            int h2 = h1 - 1;
            if (h2 >= 0 && h2 <= HMAX) {
                int pos = (cj * 681 + h2) * 6 + w1 + 3;
                if (pos < OUTPB) out[obase + pos] = y;
            }
        }
    }
}

extern "C" void kernel_launch(void* const* d_in, const int* in_sizes, int n_in,
                              void* d_out, int out_size, void* d_ws, size_t ws_size,
                              hipStream_t stream)
{
    (void)in_sizes; (void)n_in; (void)d_ws; (void)ws_size; (void)out_size;
    const float* x    = (const float*)d_in[0];   // (16, 2048, 32) f32
    const float* kers = (const float*)d_in[1];   // (74, 8, 3) f32
    float*       out  = (float*)d_out;           // (16, 2048, 74) f32

    dim3 grid(B_ * NCG);      // 128 blocks: one (batch, 4-channel group) each
    dim3 block(256);          // 8 wave32s
    size_t lds = (size_t)CPB * SIGLEN * sizeof(float);   // 33,280 B
    conv24_wmma_kernel<<<grid, block, lds, stream>>>(x, kers, out);
}
